// GTN_56057913147873
// MI455X (gfx1250) — matrix operand order repaired and checked by reading, not compiled, not run on hardware
//
#include <hip/hip_runtime.h>

typedef __attribute__((ext_vector_type(16))) __bf16       v16bf;
typedef __attribute__((ext_vector_type(8)))  float        v8f;
typedef __attribute__((ext_vector_type(4)))  unsigned int uint4v;
typedef int v4i_gcc __attribute__((vector_size(16)));

#define NN_NODES 2048
#define PLANE    (2048LL * 2048LL)
#define DREF     300
#define DPAD     384
#define NGRAPH   128

#if __has_builtin(__builtin_amdgcn_global_load_async_to_lds_b128)
#define HAVE_ASYNC_LDS 1
#else
#define HAVE_ASYNC_LDS 0
#endif

// generic -> AS(1)/AS(3) pointer via integer cast (low 32 bits of a generic
// LDS pointer are the LDS byte offset per the gfx1250 aperture rules).
// Builtin expects pointers to int4 vectors (per hipcc diagnostic).
#define GPTR(p) ((__attribute__((address_space(1))) v4i_gcc*)(unsigned long long)(p))
#define LPTR(p) ((__attribute__((address_space(3))) v4i_gcc*)(unsigned int)(unsigned long long)(p))

__device__ __forceinline__ void wait_async_lds() {
#if HAVE_ASYNC_LDS
#if __has_builtin(__builtin_amdgcn_s_wait_asynccnt)
    __builtin_amdgcn_s_wait_asynccnt(0);
#else
    asm volatile("s_wait_asynccnt 0x0" ::: "memory");
#endif
#endif
}

__device__ __forceinline__ unsigned short f2bf(float f) {
    unsigned int u = __builtin_bit_cast(unsigned int, f);
    unsigned int r = u + 0x7FFFu + ((u >> 16) & 1u);
    return (unsigned short)(r >> 16);
}

// ---------------- tiny softmax of mixing weights ----------------
// alpha layout: [set(3)][channel(2)][type(5)]
__global__ void gtn_softmax(const float* Wa, const float* Wb, const float* Wc,
                            float* alpha) {
    int t = threadIdx.x;
    if (t >= 6) return;
    int set = t >> 1, c = t & 1;
    const float* W = (set == 0) ? Wa : (set == 1 ? Wb : Wc);
    float m = -1e30f;
    for (int k = 0; k < 5; k++) m = fmaxf(m, W[c * 5 + k]);
    float ex[5], s = 0.f;
    for (int k = 0; k < 5; k++) { ex[k] = __expf(W[c * 5 + k] - m); s += ex[k]; }
    for (int k = 0; k < 5; k++) alpha[set * 10 + c * 5 + k] = ex[k] / s;
}

// ---------------- zero fill ----------------
__global__ void gtn_zero(float* p, long long n) {
    long long i = (long long)blockIdx.x * blockDim.x + threadIdx.x;
    if (i < n) p[i] = 0.f;
}

// ---------------- edge scatter-add into 2-channel dense adjacency -------------
__global__ void gtn_scatter(const int* __restrict__ idx, const int* __restrict__ attr,
                            int E, const float* __restrict__ e1,
                            const float* __restrict__ e2,
                            const float* __restrict__ alpha, int btype,
                            float* __restrict__ dst) {
    int e = blockIdx.x * blockDim.x + threadIdx.x;
    if (e >= E) return;
    int r = idx[e];
    int c = idx[E + e];
    float val = e1[attr[2 * e]] + e2[attr[2 * e + 1]];
    atomicAdd(&dst[(long long)r * NN_NODES + c], alpha[btype] * val);
    atomicAdd(&dst[PLANE + (long long)r * NN_NODES + c], alpha[5 + btype] * val);
}

__global__ void gtn_diag(float* dst, const float* alpha) {
    int i = blockIdx.x * blockDim.x + threadIdx.x;
    if (i < NN_NODES) {
        dst[(long long)i * NN_NODES + i] += alpha[4];
        dst[PLANE + (long long)i * NN_NODES + i] += alpha[9];
    }
}

// ---------------- f32 -> bf16 convert ----------------
__global__ void gtn_cvt(const float* __restrict__ s, unsigned short* __restrict__ d,
                        long long n) {
    long long i = (long long)blockIdx.x * blockDim.x + threadIdx.x;
    if (i < n) d[i] = f2bf(s[i]);
}

// ---------------- bf16 WMMA tiled GEMM: C(f32)[M,Nc] = A[M,K] @ B[K,Nc] --------
// 256 threads, 128x128 tile, K-step 32, double-buffered LDS.
// Waves 2x4; each wave computes a 64x32 region = 4x2 16x16 WMMA tiles.
union FragBF { v16bf v; uint4v q[2]; };

__device__ __forceinline__ void gtn_load_tileA(const unsigned short* __restrict__ A,
                                               int K, int tileM, int k0,
                                               unsigned short (*As)[40], int tid) {
    int row = tid >> 1;
    int seg = (tid & 1) * 16;
    const unsigned short* src = A + (long long)(tileM + row) * K + (k0 + seg);
#if HAVE_ASYNC_LDS
    __builtin_amdgcn_global_load_async_to_lds_b128(GPTR(src), LPTR(&As[row][seg]), 0, 0);
    __builtin_amdgcn_global_load_async_to_lds_b128(GPTR(src + 8), LPTR(&As[row][seg + 8]), 0, 0);
#else
    *(uint4v*)&As[row][seg]     = *(const uint4v*)(src);
    *(uint4v*)&As[row][seg + 8] = *(const uint4v*)(src + 8);
#endif
}

__device__ __forceinline__ void gtn_load_tileB(const unsigned short* __restrict__ B,
                                               int Ncols, int tileN, int k0,
                                               unsigned short (*Bt)[40], int tid) {
    // each thread owns a 4(k) x 4(n) sub-block: 4 x 8B global loads,
    // register transpose, 4 packed ds_store_b64.
    int kq = (tid >> 5) * 4;    // 0,4,...,28
    int c  = (tid & 31) * 4;    // 0..124
    unsigned long long r[4];
#pragma unroll
    for (int rr = 0; rr < 4; rr++)
        r[rr] = *(const unsigned long long*)(B + (long long)(k0 + kq + rr) * Ncols +
                                             (tileN + c));
#pragma unroll
    for (int j = 0; j < 4; j++) {
        unsigned long long w =
            ((r[0] >> (16 * j)) & 0xFFFFull) |
            (((r[1] >> (16 * j)) & 0xFFFFull) << 16) |
            (((r[2] >> (16 * j)) & 0xFFFFull) << 32) |
            (((r[3] >> (16 * j)) & 0xFFFFull) << 48);
        *(unsigned long long*)&Bt[c + j][kq] = w;
    }
}

__global__ __launch_bounds__(256) void gtn_gemm_bf16(
    const unsigned short* __restrict__ A, const unsigned short* __restrict__ B,
    float* __restrict__ C, int Ncols, int K,
    long long aStride, long long bStride, long long cStride,
    const float* __restrict__ bias, int relu) {
    const int z = blockIdx.z;
    A += (long long)z * aStride;
    B += (long long)z * bStride;
    C += (long long)z * cStride;
    const int tileM = blockIdx.y * 128;
    const int tileN = blockIdx.x * 128;

    __shared__ unsigned short As[2][128][40];  // +8 pad, rows 80B (16B aligned)
    __shared__ unsigned short Bt[2][128][40];  // transposed B tile

    const int tid  = threadIdx.x;
    const int lane = tid & 31;
    const int wave = tid >> 5;
    const int waveM = wave >> 2;   // 0..1 -> 64 rows
    const int waveN = wave & 3;    // 0..3 -> 32 cols
    const int half = lane >> 4;
    const int l15  = lane & 15;

    v8f acc[4][2];
    const v8f vzero = {0.f, 0.f, 0.f, 0.f, 0.f, 0.f, 0.f, 0.f};
#pragma unroll
    for (int i = 0; i < 4; i++)
#pragma unroll
        for (int j = 0; j < 2; j++) acc[i][j] = vzero;

    // prologue: stage tile 0 into buffer 0
    gtn_load_tileA(A, K, tileM, 0, As[0], tid);
    gtn_load_tileB(B, Ncols, tileN, 0, Bt[0], tid);
    wait_async_lds();
    __syncthreads();

    for (int k0 = 0; k0 < K; k0 += 32) {
        const int cur = (k0 >> 5) & 1;
        const int nxt = cur ^ 1;
        if (k0 + 32 < K) {   // issue next tile while computing current
            gtn_load_tileA(A, K, tileM, k0 + 32, As[nxt], tid);
            gtn_load_tileB(B, Ncols, tileN, k0 + 32, Bt[nxt], tid);
        }

        FragBF bf[2];
#pragma unroll
        for (int j = 0; j < 2; j++) {
            int n = waveN * 32 + j * 16 + l15;   // per-lane column of B
            bf[j].q[0] = *(const uint4v*)&Bt[cur][n][half * 8];  // K = half*8 + r, 16+half*8+r
            bf[j].q[1] = *(const uint4v*)&Bt[cur][n][16 + half * 8];
        }
#pragma unroll
        for (int i = 0; i < 4; i++) {
            FragBF af;
            int m = waveM * 64 + i * 16 + l15;   // per-lane row of A
            af.q[0] = *(const uint4v*)&As[cur][m][half * 8];       // K = half*8 + r
            af.q[1] = *(const uint4v*)&As[cur][m][16 + half * 8];  // K = 16+half*8+r
#pragma unroll
            for (int j = 0; j < 2; j++) {
                acc[i][j] = __builtin_amdgcn_wmma_f32_16x16x32_bf16(
                    false, af.v, false, bf[j].v, (short)0, acc[i][j], false, false);
            }
        }
        wait_async_lds();
        __syncthreads();
    }

#pragma unroll
    for (int i = 0; i < 4; i++) {
        int baseM = tileM + waveM * 64 + i * 16;
#pragma unroll
        for (int j = 0; j < 2; j++) {
            int col = tileN + waveN * 32 + j * 16 + l15;
            float bv = bias ? bias[col] : 0.f;
#pragma unroll
            for (int r = 0; r < 8; r++) {
                int rowi = baseM + r + 8 * half;   // D layout: M = r + 8*half
                float v = acc[i][j][r] + bv;
                if (relu) v = v > 0.f ? v : 0.f;
                C[(long long)rowi * Ncols + col] = v;
            }
        }
    }
}

// ---------------- row-normalize fp32 H, emit bf16 ----------------
__global__ __launch_bounds__(256) void gtn_normalize(const float* __restrict__ H,
                                                     unsigned short* __restrict__ Hbf) {
    long long base = (long long)blockIdx.y * PLANE + (long long)blockIdx.x * NN_NODES;
    __shared__ float red[256];
    float s = 0.f;
    for (int j = threadIdx.x; j < NN_NODES; j += 256) s += H[base + j];
    red[threadIdx.x] = s;
    __syncthreads();
    for (int off = 128; off > 0; off >>= 1) {
        if (threadIdx.x < off) red[threadIdx.x] += red[threadIdx.x + off];
        __syncthreads();
    }
    float deg  = red[0];
    float dinv = (deg == 0.f) ? 0.f : 1.f / deg;
    for (int j = threadIdx.x; j < NN_NODES; j += 256)
        Hbf[base + j] = f2bf(H[base + j] * dinv);
}

// -------- node features: hW = (emb1[x0]+emb2[x1]) @ W_gcn, bf16 padded --------
__global__ __launch_bounds__(256) void gtn_nodefeat(const int* __restrict__ x,
                                                    const float* __restrict__ xe1,
                                                    const float* __restrict__ xe2,
                                                    const float* __restrict__ Wg,
                                                    unsigned short* __restrict__ hWbf) {
    int i = blockIdx.x;
    __shared__ float hsh[DREF];
    int i0 = x[2 * i], i1 = x[2 * i + 1];
    for (int d = threadIdx.x; d < DREF; d += 256)
        hsh[d] = xe1[i0 * DREF + d] + xe2[i1 * DREF + d];
    __syncthreads();
    for (int o = threadIdx.x; o < DPAD; o += 256) {
        float a = 0.f;
        if (o < DREF)
            for (int k = 0; k < DREF; k++) a += hsh[k] * Wg[k * DREF + o];
        hWbf[(long long)i * DPAD + o] = f2bf(a);
    }
}

__global__ void gtn_padbias(const float* b, float* bp) {
    int d = blockIdx.x * blockDim.x + threadIdx.x;
    if (d < DPAD) bp[d] = (d < DREF) ? b[d] : 0.f;
}

// ---------------- pooling + final linear ----------------
__global__ void gtn_poolcnt(const int* batch, float* cnts) {
    int n = blockIdx.x * blockDim.x + threadIdx.x;
    if (n < NN_NODES) atomicAdd(&cnts[batch[n]], 1.f);
}

__global__ __launch_bounds__(256) void gtn_poolacc(const float* __restrict__ Xc,
                                                   const int* __restrict__ batch,
                                                   float* __restrict__ sums) {
    int n = blockIdx.x;
    int g = batch[n];
    for (int j = threadIdx.x; j < 2 * DREF; j += 256) {
        int c = j / DREF, d = j - c * DREF;
        float v = Xc[(long long)c * NN_NODES * DPAD + (long long)n * DPAD + d];
        atomicAdd(&sums[g * (2 * DREF) + j], v);
    }
}

__global__ __launch_bounds__(256) void gtn_final(const float* __restrict__ sums,
                                                 const float* __restrict__ cnts,
                                                 const float* __restrict__ Wl,
                                                 const float* __restrict__ bl,
                                                 float* __restrict__ out) {
    int g = blockIdx.x;
    __shared__ float y[2 * DREF];
    float inv = 1.f / fmaxf(cnts[g], 1.f);
    for (int j = threadIdx.x; j < 2 * DREF; j += 256)
        y[j] = sums[g * (2 * DREF) + j] * inv;
    __syncthreads();
    for (int o = threadIdx.x; o < DREF; o += 256) {
        float a = bl[o];
        for (int j = 0; j < 2 * DREF; j++) a += y[j] * Wl[o * (2 * DREF) + j];
        out[g * DREF + o] = a;
    }
}

// ================================================================
extern "C" void kernel_launch(void* const* d_in, const int* in_sizes, int n_in,
                              void* d_out, int out_size, void* d_ws, size_t ws_size,
                              hipStream_t stream) {
    (void)n_in; (void)out_size; (void)ws_size;
    const int* x        = (const int*)d_in[0];
    const int* si       = (const int*)d_in[1];
    const int* sa       = (const int*)d_in[2];
    const int* di       = (const int*)d_in[3];
    const int* da       = (const int*)d_in[4];
    const int* ti       = (const int*)d_in[5];
    const int* ta       = (const int*)d_in[6];
    const int* ai       = (const int*)d_in[7];
    const int* aa       = (const int*)d_in[8];
    const int* batch    = (const int*)d_in[9];
    const float* xe1    = (const float*)d_in[10];
    const float* xe2    = (const float*)d_in[11];
    const float* ee1    = (const float*)d_in[12];
    const float* ee2    = (const float*)d_in[13];
    const float* Wc0a   = (const float*)d_in[14];
    const float* Wc0b   = (const float*)d_in[15];
    const float* Wc1    = (const float*)d_in[16];
    const float* Wgcn   = (const float*)d_in[17];
    const float* bgcn   = (const float*)d_in[18];
    const float* Wlin   = (const float*)d_in[19];
    const float* blin   = (const float*)d_in[20];
    float* out          = (float*)d_out;

    const int Es = in_sizes[1] / 2, Ed = in_sizes[3] / 2,
              Et = in_sizes[5] / 2, Ea = in_sizes[7] / 2;

    // ---- workspace carve-out ----
    char* ws = (char*)d_ws;
    size_t off = 0;
    auto carve = [&](size_t bytes) -> char* {
        off = (off + 255) & ~(size_t)255;
        char* p = ws + off;
        off += bytes;
        return p;
    };
    float*          alpha  = (float*)carve(32 * sizeof(float));        // 3x2x5
    float*          P      = (float*)carve(2 * PLANE * 4);             // fp32 [2,N,N]
    float*          Q      = (float*)carve(2 * PLANE * 4);
    unsigned short* X      = (unsigned short*)carve(2 * PLANE * 2);    // bf16
    unsigned short* Y      = (unsigned short*)carve(2 * PLANE * 2);
    unsigned short* hWbf   = (unsigned short*)carve((size_t)NN_NODES * DPAD * 2);
    float*          biasP  = (float*)carve(DPAD * 4);
    float*          Xc     = (float*)carve(2LL * NN_NODES * DPAD * 4);
    float*          sums   = (float*)carve((size_t)NGRAPH * 2 * DREF * 4);
    float*          cnts   = (float*)carve(NGRAPH * 4);

    const long long n2 = 2 * PLANE;
    dim3 blk(256);
    dim3 zgrid((unsigned)((n2 + 255) / 256));

    // 1) softmax mixing coefficients
    gtn_softmax<<<1, 32, 0, stream>>>(Wc0a, Wc0b, Wc1, alpha);

    // 2) build comb(l0a)->P, comb(l0b)->Q
    gtn_zero<<<zgrid, blk, 0, stream>>>(P, n2);
    gtn_zero<<<zgrid, blk, 0, stream>>>(Q, n2);
    gtn_scatter<<<(Es + 255) / 256, blk, 0, stream>>>(si, sa, Es, ee1, ee2, alpha + 0, 0, P);
    gtn_scatter<<<(Ed + 255) / 256, blk, 0, stream>>>(di, da, Ed, ee1, ee2, alpha + 0, 1, P);
    gtn_scatter<<<(Et + 255) / 256, blk, 0, stream>>>(ti, ta, Et, ee1, ee2, alpha + 0, 2, P);
    gtn_scatter<<<(Ea + 255) / 256, blk, 0, stream>>>(ai, aa, Ea, ee1, ee2, alpha + 0, 3, P);
    gtn_scatter<<<(Es + 255) / 256, blk, 0, stream>>>(si, sa, Es, ee1, ee2, alpha + 10, 0, Q);
    gtn_scatter<<<(Ed + 255) / 256, blk, 0, stream>>>(di, da, Ed, ee1, ee2, alpha + 10, 1, Q);
    gtn_scatter<<<(Et + 255) / 256, blk, 0, stream>>>(ti, ta, Et, ee1, ee2, alpha + 10, 2, Q);
    gtn_scatter<<<(Ea + 255) / 256, blk, 0, stream>>>(ai, aa, Ea, ee1, ee2, alpha + 10, 3, Q);
    gtn_diag<<<NN_NODES / 256, blk, 0, stream>>>(P, alpha + 0);
    gtn_diag<<<NN_NODES / 256, blk, 0, stream>>>(Q, alpha + 10);
    gtn_cvt<<<zgrid, blk, 0, stream>>>(P, X, n2);
    gtn_cvt<<<zgrid, blk, 0, stream>>>(Q, Y, n2);

    // 3) layer 0: P = X @ Y  (per channel, WMMA bf16)
    gtn_gemm_bf16<<<dim3(16, 16, 2), blk, 0, stream>>>(
        X, Y, P, NN_NODES, NN_NODES, PLANE, PLANE, PLANE, nullptr, 0);
    gtn_normalize<<<dim3(NN_NODES, 2), blk, 0, stream>>>(P, X);

    // 4) comb(l1)->Q, layer 1: P = X @ Y
    gtn_zero<<<zgrid, blk, 0, stream>>>(Q, n2);
    gtn_scatter<<<(Es + 255) / 256, blk, 0, stream>>>(si, sa, Es, ee1, ee2, alpha + 20, 0, Q);
    gtn_scatter<<<(Ed + 255) / 256, blk, 0, stream>>>(di, da, Ed, ee1, ee2, alpha + 20, 1, Q);
    gtn_scatter<<<(Et + 255) / 256, blk, 0, stream>>>(ti, ta, Et, ee1, ee2, alpha + 20, 2, Q);
    gtn_scatter<<<(Ea + 255) / 256, blk, 0, stream>>>(ai, aa, Ea, ee1, ee2, alpha + 20, 3, Q);
    gtn_diag<<<NN_NODES / 256, blk, 0, stream>>>(Q, alpha + 20);
    gtn_cvt<<<zgrid, blk, 0, stream>>>(Q, Y, n2);
    gtn_gemm_bf16<<<dim3(16, 16, 2), blk, 0, stream>>>(
        X, Y, P, NN_NODES, NN_NODES, PLANE, PLANE, PLANE, nullptr, 0);
    gtn_normalize<<<dim3(NN_NODES, 2), blk, 0, stream>>>(P, X);

    // 5) GCN: Xc = relu(H @ hW + b)
    gtn_nodefeat<<<NN_NODES, blk, 0, stream>>>(x, xe1, xe2, Wgcn, hWbf);
    gtn_padbias<<<2, blk, 0, stream>>>(bgcn, biasP);
    gtn_gemm_bf16<<<dim3(DPAD / 128, 16, 2), blk, 0, stream>>>(
        X, hWbf, Xc, DPAD, NN_NODES, PLANE, 0, (long long)NN_NODES * DPAD, biasP, 1);

    // 6) pool + final linear
    gtn_zero<<<(NGRAPH * 2 * DREF + 255) / 256, blk, 0, stream>>>(sums, NGRAPH * 2 * DREF);
    gtn_zero<<<1, blk, 0, stream>>>(cnts, NGRAPH);
    gtn_poolcnt<<<NN_NODES / 256, blk, 0, stream>>>(batch, cnts);
    gtn_poolacc<<<NN_NODES, blk, 0, stream>>>(Xc, batch, sums);
    gtn_final<<<NGRAPH, blk, 0, stream>>>(sums, cnts, Wlin, blin, out);
}
